// DCT_2D_257698038331
// MI455X (gfx1250) — matrix operand order repaired and checked by reading, not compiled, not run on hardware
//
#include <hip/hip_runtime.h>

// ============================================================================
// 2D DCT (8,32,256,256) fp32 == Out = D · X · D^T per image, D[f][t] =
// (1/256)·cos(pi·f·(2t+1)/512).  Compute-bound (17.2 GFLOP vs ~5.5us HBM), so
// run on the fast 16-bit matrix pipe: split-precision f16x3 GEMMs with
// V_WMMA_F32_16X16X32_F16 (fp32 accum).  x = xh + xl (f16 hi + f16 residual);
// Ah*Bh + Ah*Bl + Al*Bh leaves ~2^-22 relative error per product.  The 1/256
// matrix scaling is dropped (raw cos, entries <= 1) and the exact 2^-16 is
// folded into the final store, keeping all f16 values well-scaled.
// Fused: T = X*C^T kept transposed in LDS (hi+lo planes, 264 KiB of the
// 320 KiB WGP LDS), then Out = C*T.  One workgroup (8 wave32) per image.
//
// New this round: X is pre-split into global f16 hi/lo planes by a tiny
// memory-bound pass (one conversion per element instead of 8x redundant
// in-loop conversion), removing ~all VALU from the Phase-1 inner loop.
// Gated on ws_size at runtime; the in-loop-conversion variant remains as a
// template fallback so both paths are always valid.
// ============================================================================

typedef __attribute__((ext_vector_type(16))) _Float16 v16h;
typedef __attribute__((ext_vector_type(8)))  _Float16 v8h;
typedef __attribute__((ext_vector_type(4)))  _Float16 v4h;
typedef __attribute__((ext_vector_type(8)))  float    v8f;
typedef __attribute__((ext_vector_type(8)))  float    f32x8;
typedef __attribute__((ext_vector_type(4)))  float    f32x4;

#define DCT_L 256
#define TT_P  264   // LDS pitch (halves): 528B rows -> bank step 4, 16B aligned

// C[f][t] = cos(pi * f*(2t+1)/512), split into f16 hi + f16 residual.
// f*(2t+1) < 2^24 exact in fp32; /512 exact; cospif reduces mod 2 exactly.
__global__ __launch_bounds__(256) void dct_build_c(_Float16* __restrict__ Ch,
                                                   _Float16* __restrict__ Cl) {
  const int f = blockIdx.x;
  const int t = threadIdx.x;
  const float c = cospif((float)(f * (2 * t + 1)) * (1.0f / 512.0f));
  const _Float16 h = (_Float16)c;
  Ch[f * DCT_L + t] = h;
  Cl[f * DCT_L + t] = (_Float16)(c - (float)h);
}

// Memory-bound pre-split: X (fp32) -> Xh + Xl (f16), 4 elements per thread.
__global__ __launch_bounds__(256) void dct_split_x(const float* __restrict__ X,
                                                   _Float16* __restrict__ Xh,
                                                   _Float16* __restrict__ Xl,
                                                   int n4) {
  const int idx = blockIdx.x * 256 + threadIdx.x;
  if (idx >= n4) return;
  const f32x4 x = ((const f32x4*)X)[idx];
  v4h h, l;
#pragma unroll
  for (int e = 0; e < 4; ++e) {
    h[e] = (_Float16)x[e];
    l[e] = (_Float16)(x[e] - (float)h[e]);
  }
  ((v4h*)Xh)[idx] = h;
  ((v4h*)Xl)[idx] = l;
}

__device__ __forceinline__ v16h cat8(v8h lo, v8h hi) {
  return __builtin_shufflevector(lo, hi, 0, 1, 2, 3, 4, 5, 6, 7,
                                 8, 9, 10, 11, 12, 13, 14, 15);
}

__device__ __forceinline__ v8f wmma16(v16h a, v16h b, v8f c) {
  return __builtin_amdgcn_wmma_f32_16x16x32_f16(false, a, false, b,
                                                (short)0, c, false, false);
}

template <bool PRESPLIT>
__global__ __launch_bounds__(256) void dct2d_wmma_f16x3(
    const float* __restrict__ X,        // [nimg][256][256] fp32 (fallback path)
    const _Float16* __restrict__ Xh,    // [nimg][256][256] f16 hi (presplit)
    const _Float16* __restrict__ Xl,    // [nimg][256][256] f16 lo (presplit)
    const _Float16* __restrict__ Ch,    // [256][256] cos hi
    const _Float16* __restrict__ Cl,    // [256][256] cos residual
    float* __restrict__ Out)
{
  extern __shared__ _Float16 smem[];
  _Float16* TtH = smem;                  // [256][TT_P] T^T hi
  _Float16* TtL = smem + DCT_L * TT_P;   // [256][TT_P] T^T lo
  constexpr int L = DCT_L;

  const int img  = blockIdx.x;
  const int wave = threadIdx.x >> 5;     // 0..7  -> n-strip of 32 cols
  const int lane = threadIdx.x & 31;
  const int lr   = lane & 15;
  const int lh   = lane >> 4;            // half-wave select

  const float*    __restrict__ Xi  = X  + (size_t)img * (L * L);
  const _Float16* __restrict__ XhI = Xh + (size_t)img * (L * L);
  const _Float16* __restrict__ XlI = Xl + (size_t)img * (L * L);
  float* __restrict__ Oi = Out + (size_t)img * (L * L);
  const int n0 = wave * 32;
  const v8f zero = {};

  // ---------- Phase 1: T = X * C^T ; store T^T (f16 hi/lo) to LDS ----------
  // A(m,k) = X[m][k];  B(k,n) = C[n][k] (K-contiguous)
  for (int mr = 0; mr < 4; ++mr) {
    const int m0 = mr * 64;
    v8f acc[4][2];
#pragma unroll
    for (int i = 0; i < 4; ++i)
#pragma unroll
      for (int j = 0; j < 2; ++j) acc[i][j] = zero;

    for (int k0 = 0; k0 < L; k0 += 32) {
      v16h ah[4], al[4], bh[2], bl[2];
#pragma unroll
      for (int i = 0; i < 4; ++i) {
        if constexpr (PRESPLIT) {
          const size_t off = (size_t)(m0 + 16 * i + lr) * L + k0 + 8 * lh;
          ah[i] = cat8(*(const v8h*)(XhI + off), *(const v8h*)(XhI + off + 16));
          al[i] = cat8(*(const v8h*)(XlI + off), *(const v8h*)(XlI + off + 16));
        } else {
          const float* xr = Xi + (size_t)(m0 + 16 * i + lr) * L + k0 + 8 * lh;
          f32x8 c0 = *(const f32x8*)xr;         // K = k0+8lh .. +7
          f32x8 c1 = *(const f32x8*)(xr + 16);  // K = k0+16+8lh .. +7
#pragma unroll
          for (int e = 0; e < 8; ++e) {
            const _Float16 h0 = (_Float16)c0[e];
            const _Float16 h1 = (_Float16)c1[e];
            ah[i][e]     = h0;  al[i][e]     = (_Float16)(c0[e] - (float)h0);
            ah[i][8 + e] = h1;  al[i][8 + e] = (_Float16)(c1[e] - (float)h1);
          }
        }
      }
#pragma unroll
      for (int j = 0; j < 2; ++j) {
        const size_t off = (size_t)(n0 + 16 * j + lr) * L + k0 + 8 * lh;
        bh[j] = cat8(*(const v8h*)(Ch + off), *(const v8h*)(Ch + off + 16));
        bl[j] = cat8(*(const v8h*)(Cl + off), *(const v8h*)(Cl + off + 16));
      }
#pragma unroll
      for (int i = 0; i < 4; ++i)
#pragma unroll
        for (int j = 0; j < 2; ++j) {
          acc[i][j] = wmma16(ah[i], bh[j], acc[i][j]);
          acc[i][j] = wmma16(ah[i], bl[j], acc[i][j]);
          acc[i][j] = wmma16(al[i], bh[j], acc[i][j]);
        }
    }
    // C-frag: VGPR v -> row m0+16i+v (lanes 0-15) / +8 (lanes 16-31)
#pragma unroll
    for (int i = 0; i < 4; ++i)
#pragma unroll
      for (int j = 0; j < 2; ++j)
#pragma unroll
        for (int v = 0; v < 8; ++v) {
          const int r = m0 + 16 * i + v + 8 * lh;
          const int c = n0 + 16 * j + lr;
          const float t = acc[i][j][v];
          const _Float16 th = (_Float16)t;
          TtH[c * TT_P + r] = th;
          TtL[c * TT_P + r] = (_Float16)(t - (float)th);
        }
  }

  __syncthreads();

  // ---------- Phase 2: Out = C * T ----------
  // A(m,k) = C[m][k];  B(k,n) = T[k][n] = Tt[n][k] (LDS, K-contiguous)
  for (int mr = 0; mr < 4; ++mr) {
    const int m0 = mr * 64;
    v8f acc[4][2];
#pragma unroll
    for (int i = 0; i < 4; ++i)
#pragma unroll
      for (int j = 0; j < 2; ++j) acc[i][j] = zero;

    for (int k0 = 0; k0 < L; k0 += 32) {
      v16h ah[4], al[4], bh[2], bl[2];
#pragma unroll
      for (int i = 0; i < 4; ++i) {
        const size_t off = (size_t)(m0 + 16 * i + lr) * L + k0 + 8 * lh;
        ah[i] = cat8(*(const v8h*)(Ch + off), *(const v8h*)(Ch + off + 16));
        al[i] = cat8(*(const v8h*)(Cl + off), *(const v8h*)(Cl + off + 16));
      }
#pragma unroll
      for (int j = 0; j < 2; ++j) {
        const int boff = (n0 + 16 * j + lr) * TT_P + k0 + 8 * lh;
        bh[j] = cat8(*(const v8h*)(TtH + boff), *(const v8h*)(TtH + boff + 16));
        bl[j] = cat8(*(const v8h*)(TtL + boff), *(const v8h*)(TtL + boff + 16));
      }
#pragma unroll
      for (int i = 0; i < 4; ++i)
#pragma unroll
        for (int j = 0; j < 2; ++j) {
          acc[i][j] = wmma16(ah[i], bh[j], acc[i][j]);
          acc[i][j] = wmma16(ah[i], bl[j], acc[i][j]);
          acc[i][j] = wmma16(al[i], bh[j], acc[i][j]);
        }
    }
    // store with exact 2^-16 scale (restores the two 1/256 factors)
#pragma unroll
    for (int i = 0; i < 4; ++i)
#pragma unroll
      for (int j = 0; j < 2; ++j)
#pragma unroll
        for (int v = 0; v < 8; ++v) {
          const int r = m0 + 16 * i + v + 8 * lh;
          const int c = n0 + 16 * j + lr;
          Oi[(size_t)r * L + c] = acc[i][j][v] * 0x1p-16f;
        }
  }
}

extern "C" void kernel_launch(void* const* d_in, const int* in_sizes, int n_in,
                              void* d_out, int out_size, void* d_ws, size_t ws_size,
                              hipStream_t stream) {
  (void)n_in; (void)out_size;
  const float* X = (const float*)d_in[0];
  float* Out = (float*)d_out;

  const size_t nElem = (size_t)in_sizes[0];            // 8*32*256*256
  const int nimg = (int)(nElem / (DCT_L * DCT_L));     // 256 images
  const int ldsBytes = 2 * DCT_L * TT_P * (int)sizeof(_Float16); // 264 KiB

  _Float16* Ch = (_Float16*)d_ws;                      // 128 KiB
  _Float16* Cl = Ch + DCT_L * DCT_L;                   // 128 KiB
  _Float16* Xh = Cl + DCT_L * DCT_L;                   // 2*nElem bytes
  _Float16* Xl = Xh + nElem;                           // 2*nElem bytes

  const size_t wsNeedSplit = (size_t)2 * DCT_L * DCT_L * 2 + 4 * nElem;
  const bool presplit = ws_size >= wsNeedSplit;

  (void)hipFuncSetAttribute((const void*)dct2d_wmma_f16x3<true>,
                            hipFuncAttributeMaxDynamicSharedMemorySize,
                            ldsBytes);
  (void)hipFuncSetAttribute((const void*)dct2d_wmma_f16x3<false>,
                            hipFuncAttributeMaxDynamicSharedMemorySize,
                            ldsBytes);

  dct_build_c<<<DCT_L, DCT_L, 0, stream>>>(Ch, Cl);
  if (presplit) {
    const int n4 = (int)(nElem / 4);
    dct_split_x<<<(n4 + 255) / 256, 256, 0, stream>>>(X, Xh, Xl, n4);
    dct2d_wmma_f16x3<true><<<nimg, 256, ldsBytes, stream>>>(X, Xh, Xl, Ch, Cl, Out);
  } else {
    dct2d_wmma_f16x3<false><<<nimg, 256, ldsBytes, stream>>>(X, Xh, Xl, Ch, Cl, Out);
  }
}
